// ContrastiveLoss_56229711839259
// MI455X (gfx1250) — compile-verified, matrix operand order checked by reading, not run on hardware
//
#include <hip/hip_runtime.h>
#include <math.h>

#define NROWS  8192
#define DIM    256
#define MARGIN 0.2f
#define BM     256            // block tile M
#define BN     128            // block tile N
#define KS     128            // K staging width
#define LDK    (KS + 8)       // padded LDS row in ushorts (272B = 17*16B, bank-step 4)

typedef __bf16 bf16x16 __attribute__((ext_vector_type(16)));
typedef float  f32x8   __attribute__((ext_vector_type(8)));

__device__ __forceinline__ unsigned short f32_to_bf16_rne(float f) {
  union { float f; unsigned int u; } v; v.f = f;
  unsigned int r = v.u + 0x7FFFu + ((v.u >> 16) & 1u);
  return (unsigned short)(r >> 16);
}

// ---------------- Phase 1: row norms + diagonal scores, zero accumulator ----
__global__ void diag_kernel(const float* __restrict__ im, const float* __restrict__ s,
                            float* __restrict__ im_sq, float* __restrict__ s_sq,
                            float* __restrict__ diag, double* __restrict__ acc) {
  if (blockIdx.x == 0 && threadIdx.x == 0) *acc = 0.0;
  const int wave = threadIdx.x >> 5;
  const int lane = threadIdx.x & 31;
  const int row  = blockIdx.x * 8 + wave;   // 1024 blocks * 8 waves = 8192 rows
  const float4* imr = (const float4*)(im + (size_t)row * DIM);
  const float4* sr  = (const float4*)(s  + (size_t)row * DIM);
  float aim = 0.f, as = 0.f, ad = 0.f;
#pragma unroll
  for (int t = 0; t < 2; ++t) {             // 64 float4 per row / 32 lanes
    float4 a = imr[lane + t * 32];
    float4 b = sr [lane + t * 32];
    aim += a.x*a.x + a.y*a.y + a.z*a.z + a.w*a.w;
    as  += b.x*b.x + b.y*b.y + b.z*b.z + b.w*b.w;
    float dx = a.x-b.x, dy = a.y-b.y, dz = a.z-b.z, dw = a.w-b.w;
    ad  += dx*dx + dy*dy + dz*dz + dw*dw;
  }
#pragma unroll
  for (int off = 16; off > 0; off >>= 1) {  // wave32 reduction
    aim += __shfl_xor(aim, off, 32);
    as  += __shfl_xor(as,  off, 32);
    ad  += __shfl_xor(ad,  off, 32);
  }
  if (lane == 0) {
    im_sq[row] = aim;
    s_sq[row]  = as;
    diag[row]  = -sqrtf(ad);               // scores[i,i] = -||im_i - s_i||
  }
}

// ---------------- Phase 2: fused WMMA tile + hinge epilogue -----------------
__launch_bounds__(256)
__global__ void loss_kernel(const float* __restrict__ im, const float* __restrict__ s,
                            const float* __restrict__ im_sq, const float* __restrict__ s_sq,
                            const float* __restrict__ diag, double* __restrict__ acc) {
  __shared__ unsigned short lds_im[BM * LDK];   // bf16 tile of im rows (256 x 128)
  __shared__ unsigned short lds_s [BN * LDK];   // bf16 tile of s rows  (128 x 128)
  __shared__ float lds_imsq[BM], lds_di[BM], lds_ssq[BN], lds_dj[BN];
  __shared__ float red[256];

  const int tid  = threadIdx.x;
  const int i0   = blockIdx.x * BM;
  const int j0   = blockIdx.y * BN;
  const int lane = tid & 31;
  const int wave = tid >> 5;
  const int wm   = (wave >> 1) * 64;      // 4 waves over 256 rows
  const int wn   = (wave & 1) * 64;       // 2 waves over 128 cols
  const int half = lane >> 4;
  const int l15  = lane & 15;

  lds_imsq[tid] = im_sq[i0 + tid];        // 256 threads cover BM
  lds_di[tid]   = diag[i0 + tid];
  if (tid < BN) { lds_ssq[tid] = s_sq[j0 + tid]; lds_dj[tid] = diag[j0 + tid]; }

  f32x8 c[4][4];
#pragma unroll
  for (int mi = 0; mi < 4; ++mi)
#pragma unroll
    for (int ni = 0; ni < 4; ++ni) c[mi][ni] = (f32x8)0.0f;

  union FragU { bf16x16 v; uint4 q[2]; };

  for (int ko = 0; ko < DIM; ko += KS) {
    // stage f32 -> bf16: im 256x128 (32 float4/thread), s 128x128 (16/thread)
#pragma unroll
    for (int it = 0; it < 32; ++it) {
      int idx = tid + it * 256;            // float4 index, 0..8191
      int row = idx >> 5;                  // 32 float4 per row
      int c4  = (idx & 31) << 2;
      float4 a = *(const float4*)(im + (size_t)(i0 + row) * DIM + ko + c4);
      unsigned int pa0 = (unsigned)f32_to_bf16_rne(a.x) | ((unsigned)f32_to_bf16_rne(a.y) << 16);
      unsigned int pa1 = (unsigned)f32_to_bf16_rne(a.z) | ((unsigned)f32_to_bf16_rne(a.w) << 16);
      *(uint2*)&lds_im[row * LDK + c4] = make_uint2(pa0, pa1);
    }
#pragma unroll
    for (int it = 0; it < 16; ++it) {
      int idx = tid + it * 256;            // float4 index, 0..4095
      int row = idx >> 5;
      int c4  = (idx & 31) << 2;
      float4 b = *(const float4*)(s + (size_t)(j0 + row) * DIM + ko + c4);
      unsigned int pb0 = (unsigned)f32_to_bf16_rne(b.x) | ((unsigned)f32_to_bf16_rne(b.y) << 16);
      unsigned int pb1 = (unsigned)f32_to_bf16_rne(b.z) | ((unsigned)f32_to_bf16_rne(b.w) << 16);
      *(uint2*)&lds_s[row * LDK + c4] = make_uint2(pb0, pb1);
    }
    __syncthreads();

#pragma unroll
    for (int ks = 0; ks < KS; ks += 32) {  // 4 k-steps of 16x16x32 per stage
      const int kb = ks + half * 8;        // ISA 16-bit A layout: lane-half picks K block
      FragU b[4];
#pragma unroll
      for (int ni = 0; ni < 4; ++ni) {
        const unsigned short* p = &lds_s[(wn + ni * 16 + l15) * LDK + kb];
        b[ni].q[0] = *(const uint4*)p;
        b[ni].q[1] = *(const uint4*)(p + 16);
      }
#pragma unroll
      for (int mi = 0; mi < 4; ++mi) {     // stream A frags: one live at a time
        FragU a;
        const unsigned short* p = &lds_im[(wm + mi * 16 + l15) * LDK + kb];
        a.q[0] = *(const uint4*)p;
        a.q[1] = *(const uint4*)(p + 16);
#pragma unroll
        for (int ni = 0; ni < 4; ++ni)
          c[mi][ni] = __builtin_amdgcn_wmma_f32_16x16x32_bf16(
              false, a.v, false, b[ni].v, (short)0, c[mi][ni], false, false);
      }
    }
    __syncthreads();
  }

  // Fused epilogue: d2 -> score -> hinge terms, accumulated locally
  float local = 0.f;
#pragma unroll
  for (int mi = 0; mi < 4; ++mi) {
#pragma unroll
    for (int ni = 0; ni < 4; ++ni) {
      const int jl = wn + ni * 16 + l15;   // C/D layout: n = lane&15
      const int gj = j0 + jl;
      const float ssq = lds_ssq[jl];
      const float dj  = lds_dj[jl];
#pragma unroll
      for (int v = 0; v < 8; ++v) {        // C/D layout: m = vgpr + 8*(lane>=16)
        const int il = wm + mi * 16 + v + half * 8;
        const int gi = i0 + il;
        const float dot = c[mi][ni][v];
        const float d2  = lds_imsq[il] + ssq - 2.0f * dot;
        const float score = -sqrtf(fmaxf(d2, 0.f));
        if (gi != gj) {
          local += fmaxf(MARGIN + score - lds_di[il], 0.f)   // cost_s
                 + fmaxf(MARGIN + score - dj,        0.f);   // cost_im
        }
      }
    }
  }

  red[tid] = local;
  __syncthreads();
#pragma unroll
  for (int off = 128; off > 0; off >>= 1) {
    if (tid < off) red[tid] += red[tid + off];
    __syncthreads();
  }
  if (tid == 0) atomicAdd(acc, (double)red[0]);   // global_atomic_add_f64
}

// ---------------- Phase 3: finalize -----------------------------------------
__global__ void finalize_kernel(const double* __restrict__ acc, float* __restrict__ out) {
  out[0] = (float)(*acc / ((double)NROWS * (double)NROWS));
}

extern "C" void kernel_launch(void* const* d_in, const int* in_sizes, int n_in,
                              void* d_out, int out_size, void* d_ws, size_t ws_size,
                              hipStream_t stream) {
  (void)in_sizes; (void)n_in; (void)out_size; (void)ws_size;
  const float* im = (const float*)d_in[0];
  const float* s  = (const float*)d_in[1];

  double* acc   = (double*)d_ws;
  float*  im_sq = (float*)((char*)d_ws + 256);
  float*  s_sq  = im_sq + NROWS;
  float*  diag  = s_sq  + NROWS;

  diag_kernel<<<NROWS / 8, 256, 0, stream>>>(im, s, im_sq, s_sq, diag, acc);
  dim3 grid(NROWS / BM, NROWS / BN);       // 32 x 64
  loss_kernel<<<grid, 256, 0, stream>>>(im, s, im_sq, s_sq, diag, acc);
  finalize_kernel<<<1, 1, 0, stream>>>(acc, (float*)d_out);
}
